// SlidingWindowAttentionNoOverlap_64811056497013
// MI455X (gfx1250) — matrix operand order repaired and checked by reading, not compiled, not run on hardware
//
#include <hip/hip_runtime.h>
#include <hip/hip_bf16.h>

typedef __attribute__((ext_vector_type(16))) __bf16 v16bf;
typedef __attribute__((ext_vector_type(8)))  float  v8f;
typedef __attribute__((ext_vector_type(4)))  __bf16 bf16x4;
typedef __attribute__((ext_vector_type(4)))  float  f32x4;

#define B_   4
#define S_   8192
#define H_   16
#define D_   64
#define W_   128
#define C_   (S_ / W_)     // 64 chunks
#define NK   384           // 3*W keys per chunk
#define NWAVE 8
#define NTHR (NWAVE * 32)

#define KS   392           // Kt row stride (keys), padded
#define VS   72            // V row stride (feats), padded
#define SS   388           // score row stride (f32), padded

__device__ __forceinline__ v8f zero8() {
    v8f z;
#pragma unroll
    for (int i = 0; i < 8; ++i) z[i] = 0.0f;
    return z;
}

__launch_bounds__(NTHR, 1)
__global__ void swa_noovl_kernel(const float* __restrict__ q,
                                 const float* __restrict__ k,
                                 const float* __restrict__ v,
                                 float* __restrict__ out,    // [B,S,H,D]
                                 float* __restrict__ attn)   // [B,S,H,3W]
{
    __shared__ __bf16 sKt[D_][KS];        // transposed keys: [feat][key]
    __shared__ __bf16 sV [NK][VS];        // values: [key][feat]
    __shared__ float  sP [NWAVE][16][SS]; // per-wave scores / probs

    const int tid  = threadIdx.x;
    const int lane = tid & 31;
    const int wave = tid >> 5;

    const int h = blockIdx.x % H_;
    const int c = (blockIdx.x / H_) % C_;
    const int b = blockIdx.x / (H_ * C_);

    const int s0    = c * W_;        // first query row of this chunk
    const int skey0 = s0 - W_;       // key position of ext index 0

    // ---------------- stage K^T and V into LDS (bf16, zero padded) -------------
    for (int idx = tid; idx < NK * 16; idx += NTHR) {
        const int j  = idx >> 4;          // key 0..383
        const int fd = (idx & 15) * 4;    // feature group of 4
        const int sk = skey0 + j;
        float4 kv = make_float4(0.f, 0.f, 0.f, 0.f);
        float4 vv = make_float4(0.f, 0.f, 0.f, 0.f);
        if (sk >= 0 && sk < S_) {
            const size_t off = (((size_t)b * S_ + sk) * H_ + h) * D_ + fd;
            kv = *(const float4*)(k + off);
            vv = *(const float4*)(v + off);
        }
        sKt[fd + 0][j] = (__bf16)kv.x;
        sKt[fd + 1][j] = (__bf16)kv.y;
        sKt[fd + 2][j] = (__bf16)kv.z;
        sKt[fd + 3][j] = (__bf16)kv.w;
        bf16x4 vb;
        vb[0] = (__bf16)vv.x; vb[1] = (__bf16)vv.y;
        vb[2] = (__bf16)vv.z; vb[3] = (__bf16)vv.w;
        *(bf16x4*)&sV[j][fd] = vb;
    }
    __syncthreads();

    // ---------------- QK^T: one 16-row M-tile per wave -------------------------
    const int m0      = wave * 16;
    const int arow    = lane & 15;        // A-matrix row within tile (lane layout)
    const int ahalf   = lane >> 4;
    const int ak0     = ahalf * 8;        // A k-offset base per half-wave

    // preload Q A-fragments for K = 0..31 and 32..63
    v16bf aq0, aq1;
    {
        const float* qp = q + (((size_t)b * S_ + (s0 + m0 + arow)) * H_ + h) * D_;
#pragma unroll
        for (int i = 0; i < 8; ++i) {
            aq0[i]     = (__bf16)qp[ak0 + i];
            aq0[i + 8] = (__bf16)qp[ak0 + 16 + i];
            aq1[i]     = (__bf16)qp[32 + ak0 + i];
            aq1[i + 8] = (__bf16)qp[32 + ak0 + 16 + i];
        }
    }

#pragma unroll
    for (int nt = 0; nt < 24; ++nt) {
        v8f acc = zero8();
#pragma unroll
        for (int ks = 0; ks < 2; ++ks) {
            // B fragment: lane = key col of B, elements = feature rows (K dim, table order)
            union { v16bf v; bf16x4 q4[4]; } ub;
            const int bn = nt * 16 + (lane & 15);
#pragma unroll
            for (int i = 0; i < 8; ++i) {
                ub.v[i]     = sKt[ks * 32 + ak0 + i][bn];
                ub.v[i + 8] = sKt[ks * 32 + ak0 + 16 + i][bn];
            }
            acc = __builtin_amdgcn_wmma_f32_16x16x32_bf16(
                false, (ks == 0 ? aq0 : aq1), false, ub.v, (short)0, acc, false, false);
        }
        // D layout: VGPR r -> (M=r, N=lane) lanes 0-15 ; (M=r+8) lanes 16-31
#pragma unroll
        for (int rr = 0; rr < 8; ++rr)
            sP[wave][rr + ahalf * 8][nt * 16 + (lane & 15)] = acc[rr];
    }

    // ---------------- softmax over 384 (2 lanes per row, 192 cols each) --------
    const int srow = lane & 15;
    const int cbeg = ahalf * 192;
    float* prow = &sP[wave][srow][0];

    float mx = -3.0e38f;
#pragma unroll 4
    for (int i = 0; i < 192; i += 4) {
        float4 sv = *(const float4*)(prow + cbeg + i);
        mx = fmaxf(mx, fmaxf(fmaxf(sv.x, sv.y), fmaxf(sv.z, sv.w)));
    }
    mx = fmaxf(mx, __shfl_xor(mx, 16, 32));

    float sum = 0.f;
#pragma unroll 4
    for (int i = 0; i < 192; i += 4) {
        float4 sv = *(const float4*)(prow + cbeg + i);
        sv.x = __expf(sv.x - mx);
        sv.y = __expf(sv.y - mx);
        sv.z = __expf(sv.z - mx);
        sv.w = __expf(sv.w - mx);
        *(float4*)(prow + cbeg + i) = sv;
        sum += sv.x + sv.y + sv.z + sv.w;
    }
    sum += __shfl_xor(sum, 16, 32);
    const float inv = 1.0f / sum;

    // normalize in LDS and stream attn to global (f32, coalesced, non-temporal)
    f32x4* arow_g = (f32x4*)(attn +
        (((size_t)b * S_ + (s0 + m0 + srow)) * H_ + h) * (3 * W_) + cbeg);
#pragma unroll 4
    for (int i = 0; i < 192; i += 4) {
        float4 sv = *(const float4*)(prow + cbeg + i);
        sv.x *= inv; sv.y *= inv; sv.z *= inv; sv.w *= inv;
        *(float4*)(prow + cbeg + i) = sv;
        f32x4 o;
        o[0] = sv.x; o[1] = sv.y; o[2] = sv.z; o[3] = sv.w;
        __builtin_nontemporal_store(o, arow_g + (i >> 2));  // streaming: bypass cache retention
    }

    // ---------------- PV: out[16 x 64] = P[16 x 384] @ V[384 x 64] ------------
    v8f oacc[4];
#pragma unroll
    for (int nt = 0; nt < 4; ++nt) oacc[nt] = zero8();

#pragma unroll
    for (int kt = 0; kt < 12; ++kt) {
        // A fragment from probs (convert f32 -> bf16 in regs)
        v16bf ap;
        const float* pp = &sP[wave][arow][kt * 32 + ak0];
#pragma unroll
        for (int i = 0; i < 8; ++i) {
            ap[i]     = (__bf16)pp[i];
            ap[i + 8] = (__bf16)pp[16 + i];
        }
#pragma unroll
        for (int nt = 0; nt < 4; ++nt) {
            // B fragment: lane = feature col of B, elements = key rows (K dim, table order)
            union { v16bf v; bf16x4 q4[4]; } ub;
            const int bn = nt * 16 + (lane & 15);
#pragma unroll
            for (int i = 0; i < 8; ++i) {
                ub.v[i]     = sV[kt * 32 + ak0 + i][bn];
                ub.v[i + 8] = sV[kt * 32 + ak0 + 16 + i][bn];
            }
            oacc[nt] = __builtin_amdgcn_wmma_f32_16x16x32_bf16(
                false, ap, false, ub.v, (short)0, oacc[nt], false, false);
        }
    }

    // store out (non-temporal stream): VGPR rr -> (M = m0 + rr + ahalf*8)
#pragma unroll
    for (int nt = 0; nt < 4; ++nt) {
#pragma unroll
        for (int rr = 0; rr < 8; ++rr) {
            const int mrow = m0 + rr + ahalf * 8;
            float* op = out + (((size_t)b * S_ + (s0 + mrow)) * H_ + h) * D_
                            + nt * 16 + (lane & 15);
            __builtin_nontemporal_store(oacc[nt][rr], op);
        }
    }
}

extern "C" void kernel_launch(void* const* d_in, const int* in_sizes, int n_in,
                              void* d_out, int out_size, void* d_ws, size_t ws_size,
                              hipStream_t stream) {
    const float* q = (const float*)d_in[0];
    const float* k = (const float*)d_in[1];
    const float* v = (const float*)d_in[2];
    float* out  = (float*)d_out;                                  // [B,S,H,D]
    float* attn = out + (size_t)B_ * S_ * H_ * D_;                // [B,S,H,3W]

    dim3 grid(B_ * C_ * H_);
    swa_noovl_kernel<<<grid, NTHR, 0, stream>>>(q, k, v, out, attn);
}
